// Colorizer_27702539059275
// MI455X (gfx1250) — compile-verified, matrix operand order checked by reading, not run on hardware
//
#include <hip/hip_runtime.h>
#include <hip/hip_bf16.h>
#include <math.h>

typedef __attribute__((ext_vector_type(16))) _Float16 v16h;
typedef __attribute__((ext_vector_type(8)))  _Float16 v8h;
typedef __attribute__((ext_vector_type(8)))  float    v8f;
typedef __attribute__((ext_vector_type(4)))  unsigned u32x4;
typedef __attribute__((ext_vector_type(8)))  int      i32x8;
typedef __attribute__((ext_vector_type(4)))  int      i32x4;

#define PI_F 3.14159265358979323846f

// k-index inside a 16x32 fp16 WMMA fragment for this lane/element
// (CDNA5 ISA 7.12.2: lanes 0-15 hold K chunks {0..7,16..23}, lanes 16-31 {8..15,24..31})
__device__ __forceinline__ int frag_k(int lane, int j) {
    return ((j >> 3) << 4) + ((lane >> 4) << 3) + (j & 7);
}

__device__ __forceinline__ float block_reduce128(float* red, int tid, float val) {
    red[tid] = val;
    __syncthreads();
    for (int s = 64; s > 0; s >>= 1) {
        if (tid < s) red[tid] += red[tid + s];
        __syncthreads();
    }
    float r = red[0];
    __syncthreads();
    return r;
}

// ---------------------------------------------------------------------------
// Tensor Data Mover: 2-D strided tile (f32 elements) -> LDS.
// Packs a D# per CDNA5 ISA ch.8: group0 {count, lds_addr, global_addr, type=2},
// group1 {data_size=4B, tensor_dim0/1, tile_dim0/1, tensor_dim0_stride}.
// Issued once per wave (TDM ignores EXEC); call from uniform control flow.
// ---------------------------------------------------------------------------
#if __has_builtin(__builtin_amdgcn_tensor_load_to_lds)
#define HAVE_TDM 1
#else
#define HAVE_TDM 0
#endif

__device__ __forceinline__ void tdm_load_2d_f32(unsigned lds_off, const void* gaddr,
                                                unsigned tensor_d0, unsigned tensor_d1,
                                                unsigned stride0, unsigned tile_d0,
                                                unsigned tile_d1) {
#if HAVE_TDM
    unsigned long long ga = (unsigned long long)(__UINTPTR_TYPE__)gaddr;
    u32x4 g0;
    g0[0] = 1u;                                      // count=1 (valid), user mode
    g0[1] = lds_off;                                 // lds_addr (bytes)
    g0[2] = (unsigned)(ga & 0xffffffffull);          // global_addr[31:0]
    g0[3] = (unsigned)((ga >> 32) & 0x1ffffffull) | (2u << 30);  // addr[56:32], type=2
    i32x8 g1;
    g1[0] = (int)(2u << 16);                                         // data_size=4B
    g1[1] = (int)((tensor_d0 & 0xffffu) << 16);                      // tensor_dim0 lo
    g1[2] = (int)((tensor_d0 >> 16) & 0xffffu) | (int)((tensor_d1 & 0xffffu) << 16);
    g1[3] = (int)((tensor_d1 >> 16) & 0xffffu) | (int)((tile_d0 & 0xffffu) << 16);
    g1[4] = (int)(tile_d1 & 0xffffu);                                // tile_dim1, dim2=0
    g1[5] = (int)stride0;                                            // dim0 stride lo
    g1[6] = 0;                                                       // stride hi, dim1 stride
    g1[7] = 0;
    i32x4 z4 = {0, 0, 0, 0};
#if __clang_major__ >= 23
    i32x8 z8 = {0, 0, 0, 0, 0, 0, 0, 0};
    __builtin_amdgcn_tensor_load_to_lds(g0, g1, z4, z4, z8, 0);
#else
    __builtin_amdgcn_tensor_load_to_lds(g0, g1, z4, z4, 0);
#endif
#endif
}

__device__ __forceinline__ void tdm_wait() {
#if HAVE_TDM && __has_builtin(__builtin_amdgcn_s_wait_tensorcnt)
    __builtin_amdgcn_s_wait_tensorcnt((short)0);
#endif
}

// ---------------------------------------------------------------------------
// SIFT descriptor over phase-subsampled patches.
// Patch n = (b, pi, pj): pixel (u,v) = x[b, u*K+pi, v*K+pj], K = 224/PS.
// Output: out[n*128 + c]  (== reference's (B,128,K,K) reshape, contiguous per b)
// ---------------------------------------------------------------------------
template <int PS>
__global__ __launch_bounds__(128) void sift_kernel(const float* __restrict__ x,
                                                   float* __restrict__ out) {
    constexpr int K    = 224 / PS;
    constexpr int NPIX = PS * PS;
    constexpr int BK   = 2 * (PS / 5);
    constexpr int STR  = PS / 4;
    constexpr int PAD  = BK / 4;
    const float half = BK * 0.5f;

    __shared__ float patch[NPIX];
    __shared__ float gw[PS];
    __shared__ float pooled[128];
    __shared__ float red[128];
    __shared__ float gsum_s, snorm_s;

    const int n   = blockIdx.x;
    const int b   = n / (K * K);
    const int p   = n % (K * K);
    const int pi  = p / K, pj = p % K;
    const int tid = threadIdx.x;

    const float* xb = x + b * 224 * 224;
    for (int i = tid; i < NPIX; i += 128) {
        int u = i / PS, v = i % PS;
        patch[i] = xb[(u * K + pi) * 224 + (v * K + pj)];
    }
    if (tid < PS) {
        float sig = PS * 0.70710678f;  // PS / sqrt(2)
        float t   = tid - (PS - 1) * 0.5f;
        gw[tid]   = expf(-0.5f * (t / sig) * (t / sig));
    }
    pooled[tid] = 0.f;
    __syncthreads();
    if (tid == 0) {
        float s = 0.f;
        for (int i = 0; i < PS; ++i) s += gw[i];
        gsum_s = s;
    }
    __syncthreads();
    const float inv_g2 = 1.f / (gsum_s * gsum_s);
    const float inv_h2 = 1.f / (half * half);

    for (int i = tid; i < NPIX; i += 128) {
        int y = i / PS, xq = i % PS;
        float xm  = patch[y * PS + (xq > 0 ? xq - 1 : 0)];
        float xp2 = patch[y * PS + (xq < PS - 1 ? xq + 1 : xq)];
        float ym  = patch[(y > 0 ? y - 1 : 0) * PS + xq];
        float yp2 = patch[(y < PS - 1 ? y + 1 : y) * PS + xq];
        float gx  = 0.5f * (xp2 - xm);
        float gy  = 0.5f * (yp2 - ym);
        float mag = sqrtf(gx * gx + gy * gy + 1e-10f) * gw[y] * gw[xq] * inv_g2;
        float ori = atan2f(gy, gx + 1e-10f) + 2.f * PI_F;
        float ob  = ori * (8.f / (2.f * PI_F));
        float fl  = floorf(ob);
        float w1  = ob - fl;
        int   b0  = ((int)fl) & 7;
        int   b1  = (b0 + 1) & 7;
        float m0  = (1.f - w1) * mag, m1 = w1 * mag;
        for (int wy = 0; wy < 4; ++wy) {
            int ty = y + PAD - wy * STR;
            if (ty < 0 || ty >= BK) continue;
            float wyv = half - fabsf(ty + 0.5f - half);
            for (int wx = 0; wx < 4; ++wx) {
                int tx = xq + PAD - wx * STR;
                if (tx < 0 || tx >= BK) continue;
                float wv = wyv * (half - fabsf(tx + 0.5f - half)) * inv_h2;
                atomicAdd(&pooled[b0 * 16 + wy * 4 + wx], m0 * wv);
                atomicAdd(&pooled[b1 * 16 + wy * 4 + wx], m1 * wv);
            }
        }
    }
    __syncthreads();

    float v = pooled[tid];
    float nrm = block_reduce128(red, tid, v * v);
    if (tid == 0) snorm_s = fmaxf(sqrtf(nrm), 1e-12f);
    __syncthreads();
    v = v / snorm_s;
    v = fminf(fmaxf(v, 0.f), 0.2f);
    nrm = block_reduce128(red, tid, v * v);
    if (tid == 0) snorm_s = fmaxf(sqrtf(nrm), 1e-12f);
    __syncthreads();
    v = v / snorm_s;
    float asum = block_reduce128(red, tid, fabsf(v));
    if (tid == 0) snorm_s = fmaxf(asum, 1e-12f);
    __syncthreads();
    v = sqrtf(v / snorm_s + 1e-10f);
    out[(size_t)n * 128 + tid] = v;
}

// ---------------------------------------------------------------------------
// 1x1 conv (einsum skip) as TDM-staged WMMA GEMM.
// A tile: 16 contiguous pixels x 32 channels (stride HW) -> LDS [k][m] f32 via TDM.
// B tile: 32 contiguous weights x 16 cout (stride Cin)   -> LDS [n][k] f32 via TDM.
// One wave32 per 16x16 tile of (M = B*HW) x (N = Cout); K = Cin in chunks of 32.
// ---------------------------------------------------------------------------
__global__ __launch_bounds__(32) void skip1x1_wmma(const float* __restrict__ S,
                                                   const float* __restrict__ W,
                                                   const float* __restrict__ bias,
                                                   float* __restrict__ out, int M,
                                                   int HW, int Cin, int Cout) {
    __shared__ __align__(16) float As32[16 * 32];  // [k][m]
    __shared__ __align__(16) float Bs32[16 * 32];  // [n][k]
    const int lane = threadIdx.x;
    const int m0 = blockIdx.x * 16;
    const int n0 = blockIdx.y * 16;
    const int row = lane & 15, hl = lane >> 4;

    // uniform: does this tile sit inside one image row-range?
    const int bb0 = m0 / HW;
    const int yx0 = m0 % HW;
    const bool tdm_ok = (m0 + 16 <= M) && (yx0 + 16 <= HW);

    const unsigned ldsA = (unsigned)(__UINTPTR_TYPE__)&As32[0];
    const unsigned ldsB = (unsigned)(__UINTPTR_TYPE__)&Bs32[0];

    v8f c = {};
    for (int k0 = 0; k0 < Cin; k0 += 32) {
        asm volatile("s_wait_dscnt 0" ::: "memory");  // prior LDS reads before overwrite
        if (tdm_ok) {
            // A: dim0 = 16 pixels (contiguous), dim1 = 32 channels (stride HW)
            tdm_load_2d_f32(ldsA, S + ((size_t)bb0 * Cin + k0) * HW + yx0,
                            /*td0=*/HW - yx0, /*td1=*/Cin - k0, /*stride0=*/HW,
                            /*tile0=*/16, /*tile1=*/32);
        } else {
            for (int i = lane; i < 512; i += 32) {
                int kk = i >> 4, mm = i & 15;
                int m = m0 + mm, k = k0 + kk;
                float v = 0.f;
                if (m < M) {
                    int bb = m / HW, yx = m % HW;
                    v = S[((size_t)bb * Cin + k) * HW + yx];
                }
                As32[kk * 16 + mm] = v;
            }
        }
        // B: dim0 = 32 k (contiguous), dim1 = 16 cout (stride Cin)
        tdm_load_2d_f32(ldsB, W + (size_t)n0 * Cin + k0,
                        /*td0=*/Cin - k0, /*td1=*/(unsigned)(Cout - n0),
                        /*stride0=*/Cin, /*tile0=*/32, /*tile1=*/16);
        tdm_wait();
        __syncthreads();

        v16h a, bf;
        const float* brow = &Bs32[row * 32 + hl * 8];
#pragma unroll
        for (int j = 0; j < 16; ++j) {
            int fk = frag_k(lane, j);
            a[j] = (_Float16)As32[fk * 16 + row];
            bf[j] = (_Float16)brow[(j >> 3) * 16 + (j & 7)];
        }
        c = __builtin_amdgcn_wmma_f32_16x16x32_f16(false, a, false, bf, (short)0, c,
                                                   false, false);
        __syncthreads();
    }
    const int co = n0 + row;
#pragma unroll
    for (int r = 0; r < 8; ++r) {
        int m = m0 + r + 8 * hl;
        if (m < M && co < Cout) {
            int bb = m / HW, yx = m % HW;
            out[((size_t)bb * Cout + co) * HW + yx] = c[r] + bias[co];
        }
    }
}

// ---------------------------------------------------------------------------
// ConvTranspose2d(k=4, s=2, p=1) as phase-split implicit GEMM with WMMA.
// Per-row (b,qy,qx) decomposition hoisted to LDS (no per-element division).
// Tiles staged as fp16 [row][k] / [n][k]; fragments load as 2x ds_load_b128.
// grid.x = 4 phases * ceil(B*Hin*Win/16), grid.y = ceil(Cout/16), block = 32.
// ---------------------------------------------------------------------------
__global__ __launch_bounds__(32) void convt_wmma(const float* __restrict__ A1,
                                                 const float* __restrict__ A2, int C1,
                                                 int C2, const float* __restrict__ Wt,
                                                 const float* __restrict__ bias,
                                                 float* __restrict__ out, int B,
                                                 int Hin, int Win, int Cout) {
    __shared__ __align__(16) _Float16 As[16 * 32];  // [row][k]
    __shared__ __align__(16) _Float16 Bs[16 * 32];  // [n][k]
    __shared__ int rowB[16], rowY[16], rowX[16], rowV[16];

    const int lane = threadIdx.x;
    const int Cin = C1 + C2;
    const int Kd = Cin * 4;
    const int Mphase = B * Hin * Win;
    const int mtiles = (Mphase + 15) >> 4;
    const int phase = blockIdx.x / mtiles;
    const int mt = blockIdx.x % mtiles;
    const int py = phase >> 1, px = phase & 1;
    const int m0 = mt * 16;
    const int n0 = blockIdx.y * 16;
    const int Hout = Hin * 2, Wout = Win * 2;
    const int row = lane & 15, hl = lane >> 4;

    __builtin_prefetch(Wt, 0, 1);

    // tap geometry for this phase (uniform): ky = ((py+1)&1)+2*ty, iy = qy + dy(ty)
    const int ky0 = (py + 1) & 1, kx0 = (px + 1) & 1;
    const int dy0 = (py == 0) ? 0 : 1, dx0 = (px == 0) ? 0 : 1;  // dy = dy0 - ty

    // hoist per-row index decomposition
    if (lane < 16) {
        int m = m0 + lane;
        int ok = m < Mphase;
        int mm = ok ? m : 0;
        int bb = mm / (Hin * Win);
        int rem = mm % (Hin * Win);
        rowB[lane] = bb;
        rowY[lane] = rem / Win;
        rowX[lane] = rem % Win;
        rowV[lane] = ok;
    }
    __syncthreads();

    v8f c = {};
    for (int k0 = 0; k0 < Kd; k0 += 32) {
        // activation tile (implicit im2col gather, zero-padded)
        for (int i = lane; i < 512; i += 32) {
            int r = i >> 5, kk = i & 31;
            int k = k0 + kk;
            float v = 0.f;
            if (rowV[r] && k < Kd) {
                int ci = k >> 2, tap = k & 3;
                int ty = tap >> 1, tx = tap & 1;
                int iy = rowY[r] + dy0 - ty;
                int ix = rowX[r] + dx0 - tx;
                if (iy >= 0 && ix >= 0 && iy < Hin && ix < Win) {
                    int bb = rowB[r];
                    if (ci < C1)
                        v = A1[(((size_t)bb * C1 + ci) * Hin + iy) * Win + ix];
                    else
                        v = A2[(((size_t)bb * C2 + (ci - C1)) * Hin + iy) * Win + ix];
                }
            }
            As[r * 32 + kk] = (_Float16)v;
        }
        // weight tile for this phase, stored [n][k]
        for (int i = lane; i < 512; i += 32) {
            int nn = i >> 5, kk = i & 31;
            int k = k0 + kk, co = n0 + nn;
            float v = 0.f;
            if (k < Kd && co < Cout) {
                int ci = k >> 2, tap = k & 3;
                int ty = tap >> 1, tx = tap & 1;
                int ky = ky0 + 2 * ty;
                int kx = kx0 + 2 * tx;
                v = Wt[((ci * Cout + co) * 4 + ky) * 4 + kx];
            }
            Bs[nn * 32 + kk] = (_Float16)v;
        }
        __syncthreads();

        // fragments: two aligned 16B runs per operand -> ds_load_b128
        const v8h* pa = (const v8h*)&As[row * 32 + hl * 8];
        const v8h* pb = (const v8h*)&Bs[row * 32 + hl * 8];
        v8h alo = pa[0], ahi = pa[2];  // +16 halfs = +2 v8h
        v8h blo = pb[0], bhi = pb[2];
        v16h a = __builtin_shufflevector(alo, ahi, 0, 1, 2, 3, 4, 5, 6, 7, 8, 9, 10,
                                         11, 12, 13, 14, 15);
        v16h bf = __builtin_shufflevector(blo, bhi, 0, 1, 2, 3, 4, 5, 6, 7, 8, 9, 10,
                                          11, 12, 13, 14, 15);
        c = __builtin_amdgcn_wmma_f32_16x16x32_f16(false, a, false, bf, (short)0, c,
                                                   false, false);
        __syncthreads();
    }
    const int co = n0 + row;
#pragma unroll
    for (int r = 0; r < 8; ++r) {
        int rr = r + 8 * hl;
        if (rowV[rr] && co < Cout) {
            int oy = 2 * rowY[rr] + py, ox = 2 * rowX[rr] + px;
            out[(((size_t)rowB[rr] * Cout + co) * Hout + oy) * Wout + ox] =
                c[r] + bias[co];
        }
    }
}

// ---------------------------------------------------------------------------
// GroupNorm + ReLU. One block per (n, group).
// ---------------------------------------------------------------------------
__global__ __launch_bounds__(256) void gn_relu_kernel(const float* __restrict__ in,
                                                      float* __restrict__ out,
                                                      const float* __restrict__ gamma,
                                                      const float* __restrict__ beta,
                                                      int C, int HW, int groups) {
    const int n = blockIdx.x / groups;
    const int g = blockIdx.x % groups;
    const int cpg = C / groups;
    const int cnt = cpg * HW;
    const float* base = in + ((size_t)n * C + g * cpg) * HW;
    float s = 0.f, s2 = 0.f;
    for (int i = threadIdx.x; i < cnt; i += 256) {
        float v = base[i];
        s += v;
        s2 += v * v;
    }
    __shared__ float rs[256], rs2[256];
    rs[threadIdx.x] = s;
    rs2[threadIdx.x] = s2;
    __syncthreads();
    for (int st = 128; st > 0; st >>= 1) {
        if (threadIdx.x < st) {
            rs[threadIdx.x] += rs[threadIdx.x + st];
            rs2[threadIdx.x] += rs2[threadIdx.x + st];
        }
        __syncthreads();
    }
    const float mean = rs[0] / cnt;
    const float var = rs2[0] / cnt - mean * mean;
    const float inv = rsqrtf(var + 1e-5f);
    float* ob = out + ((size_t)n * C + g * cpg) * HW;
    for (int i = threadIdx.x; i < cnt; i += 256) {
        int c = g * cpg + i / HW;
        float v = (base[i] - mean) * inv * gamma[c] + beta[c];
        ob[i] = fmaxf(v, 0.f);
    }
}

// ---------------------------------------------------------------------------
// Joint bilateral blur 5x5 (sigma_color=0.1 L1, sigma_space=1.5, reflect pad)
// with both guide channels == x, fused with tanh * [0.436, 0.615] -> d_out.
// ---------------------------------------------------------------------------
__global__ __launch_bounds__(256) void jbf_tanh_kernel(const float* __restrict__ h,
                                                       const float* __restrict__ x,
                                                       float* __restrict__ out, int B) {
    const int idx = blockIdx.x * 256 + threadIdx.x;
    const int total = B * 224 * 224;
    if (idx >= total) return;
    const int b = idx / (224 * 224);
    const int rem = idx % (224 * 224);
    const int y = rem / 224, xx = rem % 224;

    float g[5];
    float gs = 0.f;
#pragma unroll
    for (int i = 0; i < 5; ++i) {
        float t = (i - 2) / 1.5f;
        g[i] = expf(-0.5f * t * t);
        gs += g[i];
    }
#pragma unroll
    for (int i = 0; i < 5; ++i) g[i] /= gs;

    const float* xb = x + (size_t)b * 224 * 224;
    const float xc = xb[y * 224 + xx];
    const float* h0 = h + ((size_t)b * 2 + 0) * 224 * 224;
    const float* h1 = h + ((size_t)b * 2 + 1) * 224 * 224;
    float a0 = 0.f, a1 = 0.f, ks = 0.f;
#pragma unroll
    for (int i = 0; i < 5; ++i) {
        int ry = y + i - 2;
        ry = ry < 0 ? -ry : (ry > 223 ? 446 - ry : ry);
#pragma unroll
        for (int j = 0; j < 5; ++j) {
            int rx = xx + j - 2;
            rx = rx < 0 ? -rx : (rx > 223 ? 446 - rx : rx);
            float xw = xb[ry * 224 + rx];
            float ad = 2.f * fabsf(xw - xc);  // L1 over 2 identical guide channels
            float k = g[i] * g[j] * expf(-50.f * ad * ad);
            a0 += h0[ry * 224 + rx] * k;
            a1 += h1[ry * 224 + rx] * k;
            ks += k;
        }
    }
    float* ob = out + (size_t)b * 2 * 224 * 224;
    ob[y * 224 + xx] = tanhf(a0 / ks) * 0.436f;
    ob[224 * 224 + y * 224 + xx] = tanhf(a1 / ks) * 0.615f;
}

// ---------------------------------------------------------------------------

extern "C" void kernel_launch(void* const* d_in, const int* in_sizes, int n_in,
                              void* d_out, int out_size, void* d_ws, size_t ws_size,
                              hipStream_t stream) {
    const float* x   = (const float*)d_in[0];
    const float* d1w = (const float*)d_in[1];
    const float* d1b = (const float*)d_in[2];
    const float* g1w = (const float*)d_in[3];
    const float* g1b = (const float*)d_in[4];
    const float* s1w = (const float*)d_in[5];
    const float* s1b = (const float*)d_in[6];
    const float* d2w = (const float*)d_in[7];
    const float* d2b = (const float*)d_in[8];
    const float* g2w = (const float*)d_in[9];
    const float* g2b = (const float*)d_in[10];
    const float* s2w = (const float*)d_in[11];
    const float* s2b = (const float*)d_in[12];
    const float* d3w = (const float*)d_in[13];
    const float* d3b = (const float*)d_in[14];
    const float* g3w = (const float*)d_in[15];
    const float* g3b = (const float*)d_in[16];
    const float* d4w = (const float*)d_in[17];
    const float* d4b = (const float*)d_in[18];
    const float* g4w = (const float*)d_in[19];
    const float* g4b = (const float*)d_in[20];
    const float* fw  = (const float*)d_in[21];
    const float* fb  = (const float*)d_in[22];
    float* outp = (float*)d_out;

    const int B = in_sizes[0] / (224 * 224);

    float* ws = (float*)d_ws;
    size_t off = 0;
    float* s7   = ws + off; off += (size_t)B * 128 * 49;
    float* s14  = ws + off; off += (size_t)B * 128 * 196;
    float* s28  = ws + off; off += (size_t)B * 128 * 784;
    float* sk1  = ws + off; off += (size_t)B * 32 * 196;
    float* sk2  = ws + off; off += (size_t)B * 16 * 784;
    float* pre1 = ws + off; off += (size_t)B * 32 * 196;
    float* h1   = ws + off; off += (size_t)B * 32 * 196;
    float* pre2 = ws + off; off += (size_t)B * 16 * 784;
    float* h2   = ws + off; off += (size_t)B * 16 * 784;
    float* pre3 = ws + off; off += (size_t)B * 8 * 3136;
    float* h3   = ws + off; off += (size_t)B * 8 * 3136;
    float* pre4 = ws + off; off += (size_t)B * 4 * 12544;
    float* h4   = ws + off; off += (size_t)B * 4 * 12544;
    float* pre5 = ws + off; off += (size_t)B * 2 * 50176;

    // --- SIFT pyramids ---
    sift_kernel<32><<<B * 49, 128, 0, stream>>>(x, s7);
    sift_kernel<16><<<B * 196, 128, 0, stream>>>(x, s14);
    sift_kernel<8><<<B * 784, 128, 0, stream>>>(x, s28);

    // --- stage 1: convT(128->32) @7->14, GN(8), ReLU ---
    {
        int mt = (B * 7 * 7 + 15) / 16;
        convt_wmma<<<dim3(4 * mt, 2), 32, 0, stream>>>(s7, nullptr, 128, 0, d1w, d1b,
                                                       pre1, B, 7, 7, 32);
        gn_relu_kernel<<<B * 8, 256, 0, stream>>>(pre1, h1, g1w, g1b, 32, 14 * 14, 8);
    }
    // skip 1: 128->32 1x1 on s14 (TDM-staged WMMA)
    {
        int M = B * 196;
        skip1x1_wmma<<<dim3((M + 15) / 16, 2), 32, 0, stream>>>(s14, s1w, s1b, sk1, M,
                                                                196, 128, 32);
    }
    // --- stage 2: convT(64->16) @14->28, GN(4), ReLU ---
    {
        int mt = (B * 14 * 14 + 15) / 16;
        convt_wmma<<<dim3(4 * mt, 1), 32, 0, stream>>>(h1, sk1, 32, 32, d2w, d2b, pre2,
                                                       B, 14, 14, 16);
        gn_relu_kernel<<<B * 4, 256, 0, stream>>>(pre2, h2, g2w, g2b, 16, 28 * 28, 4);
    }
    // skip 2: 128->16 1x1 on s28 (TDM-staged WMMA)
    {
        int M = B * 784;
        skip1x1_wmma<<<dim3((M + 15) / 16, 1), 32, 0, stream>>>(s28, s2w, s2b, sk2, M,
                                                                784, 128, 16);
    }
    // --- stage 3: convT(32->8) @28->56, GN(2), ReLU ---
    {
        int mt = (B * 28 * 28 + 15) / 16;
        convt_wmma<<<dim3(4 * mt, 1), 32, 0, stream>>>(h2, sk2, 16, 16, d3w, d3b, pre3,
                                                       B, 28, 28, 8);
        gn_relu_kernel<<<B * 2, 256, 0, stream>>>(pre3, h3, g3w, g3b, 8, 56 * 56, 2);
    }
    // --- stage 4: convT(8->4) @56->112, GN(1), ReLU ---
    {
        int mt = (B * 56 * 56 + 15) / 16;
        convt_wmma<<<dim3(4 * mt, 1), 32, 0, stream>>>(h3, nullptr, 8, 0, d4w, d4b,
                                                       pre4, B, 56, 56, 4);
        gn_relu_kernel<<<B * 1, 256, 0, stream>>>(pre4, h4, g4w, g4b, 4, 112 * 112, 1);
    }
    // --- final: convT(4->2) @112->224 ---
    {
        int mt = (B * 112 * 112 + 15) / 16;
        convt_wmma<<<dim3(4 * mt, 1), 32, 0, stream>>>(h4, nullptr, 4, 0, fw, fb, pre5,
                                                       B, 112, 112, 2);
    }
    // --- joint bilateral blur + tanh*scale -> d_out ---
    {
        int total = B * 224 * 224;
        jbf_tanh_kernel<<<(total + 255) / 256, 256, 0, stream>>>(pre5, x, outp, B);
    }
}
